// SageNet_73143293051011
// MI455X (gfx1250) — matrix operand order repaired and checked by reading, not compiled, not run on hardware
//
#include <hip/hip_runtime.h>
#include <hip/hip_bf16.h>
#include <math.h>

// ---------------- problem constants (match reference) ----------------
#define BATCH   4096
#define NTOT    500
#define INCH    8
#define NSUB    77
#define NEDGE   400
#define OUTCH   2
#define ROWS    (BATCH * NSUB)        // 315392, divisible by 128

typedef __attribute__((ext_vector_type(16))) __bf16 bf16x16;
typedef __attribute__((ext_vector_type(8)))  float  f32x8;

// =====================================================================
// k_meta: counts -> 1/(cnt+1), CSR (offsets + src lists) for dst nodes
// =====================================================================
__global__ void k_meta(const int* __restrict__ ei, float* __restrict__ cnt_inv,
                       int* __restrict__ off, int* __restrict__ srcs)
{
    __shared__ int cnt[NSUB];
    __shared__ int cur[NSUB];
    const int* src = ei;
    const int* dst = ei + NEDGE;
    int t = threadIdx.x;
    if (t < NSUB) {
        int c = 0;
        for (int e = 0; e < NEDGE; e++)
            if (dst[e] == t && src[e] != dst[e]) c++;
        cnt[t] = c;
        cnt_inv[t] = 1.0f / (float)(c + 1);
    }
    __syncthreads();
    if (t == 0) {
        int run = 0;
        for (int j = 0; j < NSUB; j++) { off[j] = run; cur[j] = run; run += cnt[j]; }
        off[NSUB] = run;
        for (int e = 0; e < NEDGE; e++)
            if (src[e] != dst[e]) srcs[cur[dst[e]]++] = src[e];
    }
}

// =====================================================================
// k_cvt_t: fp32 W[K][N] -> bf16 Wt[N][K] (transpose during convert)
// grid(K/32, N/32), 256 threads.
// =====================================================================
__global__ void k_cvt_t(const float* __restrict__ in, __bf16* __restrict__ out,
                        int K, int N)
{
    __shared__ float t[32][33];
    const int kb = blockIdx.x * 32;
    const int nb = blockIdx.y * 32;
    const int tx = threadIdx.x & 31;
    const int ty = threadIdx.x >> 5;          // 0..7
    for (int i = ty; i < 32; i += 8)
        t[i][tx] = in[(size_t)(kb + i) * N + nb + tx];
    __syncthreads();
    for (int i = ty; i < 32; i += 8)
        out[(size_t)(nb + i) * K + kb + tx] = (__bf16)t[tx][i];
}

// =====================================================================
// k_layer1: per batch element: gather x[:, n_id, :], sage_conv(W1,b1),
// L2-normalize + relu, then emit concat2 = [act1 | aggr(act1)] in bf16.
// =====================================================================
__global__ void k_layer1(const float* __restrict__ x, const int* __restrict__ n_id,
                         const float* __restrict__ W1, const float* __restrict__ b1,
                         const float* __restrict__ cnt_inv, const int* __restrict__ off,
                         const int* __restrict__ srcs, __bf16* __restrict__ cc2)
{
    __shared__ float xt[NSUB][INCH];
    __shared__ float ag[NSUB][INCH];
    __shared__ float act[NSUB][64];
    __shared__ float w1[16 * 64];
    __shared__ float bb[64];
    const int tid = threadIdx.x;
    const int b   = blockIdx.x;

    for (int i = tid; i < 16 * 64; i += blockDim.x) w1[i] = W1[i];
    for (int i = tid; i < 64; i += blockDim.x)      bb[i] = b1[i];
    for (int i = tid; i < NSUB * INCH; i += blockDim.x) {
        int j = i / INCH, c = i % INCH;
        xt[j][c] = x[(size_t)b * NTOT * INCH + (size_t)n_id[j] * INCH + c];
    }
    __syncthreads();

    for (int i = tid; i < NSUB * INCH; i += blockDim.x) {
        int j = i / INCH, c = i % INCH;
        float s = xt[j][c];
        for (int e = off[j]; e < off[j + 1]; e++) s += xt[srcs[e]][c];
        ag[j][c] = s * cnt_inv[j];
    }
    __syncthreads();

    for (int i = tid; i < NSUB * 64; i += blockDim.x) {
        int j = i / 64, o = i % 64;
        float s = bb[o];
        for (int c = 0; c < INCH; c++) {
            s += xt[j][c] * w1[c * 64 + o];
            s += ag[j][c] * w1[(INCH + c) * 64 + o];
        }
        act[j][o] = s;
    }
    __syncthreads();

    if (tid < NSUB) {
        float ss = 0.f;
        for (int o = 0; o < 64; o++) ss += act[tid][o] * act[tid][o];
        float sc = 1.0f / fmaxf(sqrtf(ss), 1e-12f);
        for (int o = 0; o < 64; o++) act[tid][o] = fmaxf(act[tid][o] * sc, 0.f);
    }
    __syncthreads();

    for (int i = tid; i < NSUB * 64; i += blockDim.x) {
        int j = i / 64, o = i % 64;
        float a = act[j][o];
        float s = a;
        for (int e = off[j]; e < off[j + 1]; e++) s += act[srcs[e]][o];
        size_t row = (size_t)b * NSUB + j;
        cc2[row * 128 + o]      = (__bf16)a;
        cc2[row * 128 + 64 + o] = (__bf16)(s * cnt_inv[j]);
    }
}

// =====================================================================
// k_gemm_bf16: C[M,N] = A[M,K] @ W[K,N] + bias
//   A: bf16 row-major [M][K];  Wt: bf16 PRE-TRANSPOSED [N][K]
// 128x128 tile / 256-thread block (8 waves = 2M x 4N), 16x16x32 WMMA,
// 8 accumulators per wave, double-buffered LDS, one barrier per K-step.
// flags bit0 = relu, bit1 = bf16 output (else fp32).
// M%128==0, N%128==0, K%32==0 guaranteed by caller.
// =====================================================================
#define LDA 40   // padded LDS row (bf16 elems): 80B stride, 16B-aligned, no 4-way bank conflict

__global__ void __launch_bounds__(256)
k_gemm_bf16(const __bf16* __restrict__ A, const __bf16* __restrict__ Wt,
            const float* __restrict__ bias, void* __restrict__ outp,
            int M, int N, int K, int flags)
{
    __shared__ __align__(16) __bf16 sA[2][128][LDA];   // [m][k]
    __shared__ __align__(16) __bf16 sB[2][128][LDA];   // [n][k]

    const int tid  = threadIdx.x;
    const int lane = tid & 31;
    const int wave = tid >> 5;
    const int wm   = wave >> 2;          // 0..1 : 64-row slab
    const int wn   = wave & 3;           // 0..3 : 32-col slab
    const int h    = lane >> 4;          // lane half
    const int l16  = lane & 15;
    const int rowBase = blockIdx.y * 128;
    const int colBase = blockIdx.x * 128;

    // cooperative loads: 2 threads per row, 16 bf16 (2 x uint4) each
    const int ldRow = tid >> 1;          // 0..127
    const int ldCol = (tid & 1) * 16;    // 0 / 16

    const __bf16* gA = A  + (size_t)(rowBase + ldRow) * K + ldCol;
    const __bf16* gB = Wt + (size_t)(colBase + ldRow) * K + ldCol;

    f32x8 zero = {};
    f32x8 acc[4][2];
#pragma unroll
    for (int mi = 0; mi < 4; mi++)
#pragma unroll
        for (int ni = 0; ni < 2; ni++) acc[mi][ni] = zero;

    // prologue: load K-tile 0 into buffer 0
    uint4 ra0 = *(const uint4*)(gA);
    uint4 ra1 = *(const uint4*)(gA + 8);
    uint4 rb0 = *(const uint4*)(gB);
    uint4 rb1 = *(const uint4*)(gB + 8);
    *(uint4*)&sA[0][ldRow][ldCol]     = ra0;
    *(uint4*)&sA[0][ldRow][ldCol + 8] = ra1;
    *(uint4*)&sB[0][ldRow][ldCol]     = rb0;
    *(uint4*)&sB[0][ldRow][ldCol + 8] = rb1;
    __syncthreads();

    const int nK = K >> 5;
    for (int kt = 0; kt < nK; kt++) {
        const int cur = kt & 1;
        // prefetch next K-tile into registers (latency hidden behind WMMAs)
        if (kt + 1 < nK) {
            const __bf16* pA = gA + (size_t)(kt + 1) * 32;
            const __bf16* pB = gB + (size_t)(kt + 1) * 32;
            ra0 = *(const uint4*)(pA);
            ra1 = *(const uint4*)(pA + 8);
            rb0 = *(const uint4*)(pB);
            rb1 = *(const uint4*)(pB + 8);
        }

        // fragments per ISA 7.12.2 layouts:
        // A (16-bit 16x32): elems 0..7 -> K=8h+e ; elems 8..15 -> K=16+8h+e
        // B (col n = l16) : elems 0..7 -> K=8h+e ; elems 8..15 -> K=16+8h+e  (rows of pre-transposed sB)
        union F { bf16x16 v; uint4 q[2]; };
        F fa[4], fb[2];
#pragma unroll
        for (int mi = 0; mi < 4; mi++) {
            const __bf16* p = &sA[cur][wm * 64 + mi * 16 + l16][0];
            fa[mi].q[0] = *(const uint4*)(p + 8 * h);
            fa[mi].q[1] = *(const uint4*)(p + 16 + 8 * h);
        }
#pragma unroll
        for (int ni = 0; ni < 2; ni++) {
            const __bf16* p = &sB[cur][wn * 32 + ni * 16 + l16][0];
            fb[ni].q[0] = *(const uint4*)(p + 8 * h);
            fb[ni].q[1] = *(const uint4*)(p + 16 + 8 * h);
        }

#pragma unroll
        for (int mi = 0; mi < 4; mi++)
#pragma unroll
            for (int ni = 0; ni < 2; ni++)
                acc[mi][ni] = __builtin_amdgcn_wmma_f32_16x16x32_bf16(
                    false, fa[mi].v, false, fb[ni].v, (short)0, acc[mi][ni],
                    false, false);

        // write prefetched tile into the other buffer (safe: its readers
        // finished at the barrier that ended iteration kt-1)
        if (kt + 1 < nK) {
            const int nxt = cur ^ 1;
            *(uint4*)&sA[nxt][ldRow][ldCol]     = ra0;
            *(uint4*)&sA[nxt][ldRow][ldCol + 8] = ra1;
            *(uint4*)&sB[nxt][ldRow][ldCol]     = rb0;
            *(uint4*)&sB[nxt][ldRow][ldCol + 8] = rb1;
        }
        __syncthreads();
    }

    // epilogue: C/D layout -> lane l16 = N, VGPR r -> M = r + 8h
#pragma unroll
    for (int ni = 0; ni < 2; ni++) {
        const int col = colBase + wn * 32 + ni * 16 + l16;
        const float bi = bias ? bias[col] : 0.f;
#pragma unroll
        for (int mi = 0; mi < 4; mi++) {
#pragma unroll
            for (int r = 0; r < 8; r++) {
                const int row = rowBase + wm * 64 + mi * 16 + r + 8 * h;
                float v = acc[mi][ni][r] + bi;
                if (flags & 1) v = fmaxf(v, 0.f);
                if (flags & 2) ((__bf16*)outp)[(size_t)row * N + col] = (__bf16)v;
                else           ((float*)outp)[(size_t)row * N + col]  = v;
            }
        }
    }
}

// =====================================================================
// k_rownorm_relu: per-row L2 normalize + relu, bf16 in/out (in-place safe)
// =====================================================================
__global__ void k_rownorm_relu(const __bf16* __restrict__ in, __bf16* __restrict__ out,
                               int C, int inStride, int outStride)
{
    __shared__ float red[256];
    const size_t row = blockIdx.x;
    const int tid = threadIdx.x;
    const int i0 = tid, i1 = tid + 256;
    float v0 = 0.f, v1 = 0.f;
    if (i0 < C) v0 = (float)in[row * inStride + i0];
    if (i1 < C) v1 = (float)in[row * inStride + i1];
    red[tid] = v0 * v0 + v1 * v1;
    __syncthreads();
    for (int s = 128; s > 0; s >>= 1) {
        if (tid < s) red[tid] += red[tid + s];
        __syncthreads();
    }
    const float sc = 1.0f / fmaxf(sqrtf(red[0]), 1e-12f);
    if (i0 < C) out[row * outStride + i0] = (__bf16)fmaxf(v0 * sc, 0.f);
    if (i1 < C) out[row * outStride + i1] = (__bf16)fmaxf(v1 * sc, 0.f);
}

// =====================================================================
// k_aggr: fill second half of a [x | aggr] concat buffer using CSR.
// =====================================================================
__global__ void k_aggr(__bf16* __restrict__ cc, int C,
                       const int* __restrict__ off, const int* __restrict__ srcs,
                       const float* __restrict__ cnt_inv)
{
    const int b = blockIdx.x / NSUB;
    const int j = blockIdx.x % NSUB;
    const int beg = off[j], end = off[j + 1];
    const float inv = cnt_inv[j];
    const size_t rowSelf = ((size_t)b * NSUB + j) * (2 * C);
    for (int c = threadIdx.x; c < C; c += blockDim.x) {
        float s = (float)cc[rowSelf + c];
        for (int e = beg; e < end; e++) {
            size_t rs = ((size_t)b * NSUB + srcs[e]) * (2 * C);
            s += (float)cc[rs + c];
        }
        cc[rowSelf + C + c] = (__bf16)(s * inv);
    }
}

// =====================================================================
// k_fc: small scalar fp32 FC: out[m,n] = act(in[m,:] @ W[:,n] + b[n])
// =====================================================================
__global__ void k_fc(const float* __restrict__ in, const float* __restrict__ W,
                     const float* __restrict__ bias, float* __restrict__ out,
                     int M, int K, int N, int relu)
{
    const int n = blockIdx.x * blockDim.x + threadIdx.x;
    const int m = blockIdx.y;
    if (n >= N) return;
    float s = bias[n];
    const float* row = in + (size_t)m * K;
    for (int k = 0; k < K; k++) s += row[k] * W[(size_t)k * N + n];
    if (relu) s = fmaxf(s, 0.f);
    out[(size_t)m * N + n] = s;
}

// =====================================================================
// launch
// =====================================================================
extern "C" void kernel_launch(void* const* d_in, const int* in_sizes, int n_in,
                              void* d_out, int out_size, void* d_ws, size_t ws_size,
                              hipStream_t stream)
{
    const float* x    = (const float*)d_in[0];
    const int*   n_id = (const int*)d_in[1];
    const int*   ei   = (const int*)d_in[2];
    const float* W1   = (const float*)d_in[3];
    const float* b1   = (const float*)d_in[4];
    const float* W2   = (const float*)d_in[5];
    const float* b2   = (const float*)d_in[6];
    const float* W3   = (const float*)d_in[7];
    const float* b3   = (const float*)d_in[8];
    const float* L1w  = (const float*)d_in[9];
    const float* L1b  = (const float*)d_in[10];
    const float* L2w  = (const float*)d_in[11];
    const float* L2b  = (const float*)d_in[12];
    const float* L3w  = (const float*)d_in[13];
    const float* L3b  = (const float*)d_in[14];

    char* ws = (char*)d_ws;
    size_t cur = 0;
    auto alloc = [&](size_t bytes) -> char* {
        char* p = ws + cur;
        cur = (cur + bytes + 255) & ~(size_t)255;
        return p;
    };

    float*   cnt_inv = (float*)alloc(NSUB * 4);
    int*     csr_off = (int*)alloc((NSUB + 1) * 4);
    int*     csr_src = (int*)alloc(NEDGE * 4);
    __bf16*  W2t     = (__bf16*)alloc((size_t)128 * 256 * 2);      // [256][128]
    __bf16*  W3t     = (__bf16*)alloc((size_t)512 * 512 * 2);      // [512][512]
    __bf16*  L1wt    = (__bf16*)alloc((size_t)39424 * 1024 * 2);   // [1024][39424]
    // Region 1: CC2 (rows x 128 bf16) then reused for H3/X3 (rows x 512 bf16)
    char*    R1      = alloc((size_t)ROWS * 512 * 2);
    // Region 2: H2 (rows x 256 bf16) then reused for H4 (fp32) + A5 (fp32)
    char*    R2      = alloc((size_t)ROWS * 256 * 2);
    // Region 3: CC3 (rows x 512 bf16)
    char*    R3      = alloc((size_t)ROWS * 512 * 2);

    __bf16* CC2 = (__bf16*)R1;
    __bf16* H3  = (__bf16*)R1;          // also the flattened (4096 x 39424) L1 input
    __bf16* H2  = (__bf16*)R2;
    float*  H4  = (float*)R2;
    float*  A5  = (float*)(R2 + (((size_t)BATCH * 1024 * 4 + 255) & ~(size_t)255));
    __bf16* CC3 = (__bf16*)R3;

    // graph metadata
    k_meta<<<1, 128, 0, stream>>>(ei, cnt_inv, csr_off, csr_src);
    // weight conversions: fp32 [K][N] -> bf16 [N][K]
    k_cvt_t<<<dim3(128 / 32, 256 / 32),    256, 0, stream>>>(W2,  W2t,  128,   256);
    k_cvt_t<<<dim3(512 / 32, 512 / 32),    256, 0, stream>>>(W3,  W3t,  512,   512);
    k_cvt_t<<<dim3(39424 / 32, 1024 / 32), 256, 0, stream>>>(L1w, L1wt, 39424, 1024);
    // layer 1 (scalar, tiny) -> CC2 = [act1 | aggr1] bf16
    k_layer1<<<BATCH, 128, 0, stream>>>(x, n_id, W1, b1, cnt_inv, csr_off, csr_src, CC2);
    // layer 2 GEMM: (rows,128)@(128,256) -> H2 bf16 (pre-norm, bias added)
    k_gemm_bf16<<<dim3(256 / 128, ROWS / 128), 256, 0, stream>>>(
        CC2, W2t, b2, H2, ROWS, 256, 128, /*bf16 out*/2);
    // norm+relu -> first half of CC3; then aggregation -> second half
    k_rownorm_relu<<<ROWS, 256, 0, stream>>>(H2, CC3, 256, 256, 512);
    k_aggr<<<ROWS, 256, 0, stream>>>(CC3, 256, csr_off, csr_src, cnt_inv);
    // layer 3 GEMM: (rows,512)@(512,512) -> H3 bf16
    k_gemm_bf16<<<dim3(512 / 128, ROWS / 128), 256, 0, stream>>>(
        CC3, W3t, b3, H3, ROWS, 512, 512, 2);
    // norm+relu in place -> H3 is now the flattened (4096, 39424) bf16 input
    k_rownorm_relu<<<ROWS, 256, 0, stream>>>(H3, H3, 512, 512, 512);
    // L1 FC GEMM: (4096, 39424)@(39424, 1024) -> H4 fp32, fused relu
    k_gemm_bf16<<<dim3(1024 / 128, BATCH / 128), 256, 0, stream>>>(
        H3, L1wt, L1b, H4, BATCH, 1024, 39424, /*relu*/1);
    // L2 + L3 scalar FCs
    k_fc<<<dim3((218 + 63) / 64, BATCH), 64, 0, stream>>>(H4, L2w, L2b, A5,
                                                          BATCH, 1024, 218, 1);
    k_fc<<<dim3(1, BATCH), 64, 0, stream>>>(A5, L3w, L3b, (float*)d_out,
                                            BATCH, 218, OUTCH, 0);
}
